// SoftmaxNeigLoss_47124381171853
// MI455X (gfx1250) — matrix operand order repaired and checked by reading, not compiled, not run on hardware
//
#include <hip/hip_runtime.h>
#include <hip/hip_bf16.h>
#include <stdint.h>

// ---------------------------------------------------------------------------
// SoftmaxNeigLoss on MI455X (gfx1250)
//   G = X X^T via split-bf16 WMMA (hi*hi + hi*lo + lo*hi ~ fp32 accuracy)
//   d = sqrt(max(sq_i + sq_j - 2 g, 1e-12)), e = exp(50*(1-d))
//   row-sums of masked e -> loss; global masked d sums -> pos_d / neg_d
// ---------------------------------------------------------------------------

#define N_PTS   8192
#define DIMS    128
#define ALPHA_C 50.0f
#define BASE_C  1.0f
#define TILE    128
#define LDS_PITCH 136   // 128 + 8 pad elements (16B) -> 272B row stride, no bank conflicts

typedef __attribute__((ext_vector_type(16))) __bf16 v16bf;
typedef __attribute__((ext_vector_type(8)))  float  v8f;

union FragBF16 { v16bf v; uint4 q[2]; };

__device__ __forceinline__ v8f wmma_bf16(v16bf a, v16bf b, v8f c) {
    // D = A(16x32 bf16) * B(32x16 bf16) + C(16x16 f32)
    return __builtin_amdgcn_wmma_f32_16x16x32_bf16(
        /*neg_a=*/false, a, /*neg_b=*/false, b,
        /*c_mod=*/(short)0, c, /*reuse_a=*/false, /*reuse_b=*/false);
}

__device__ __forceinline__ unsigned short f32_to_bf16_rne(float f) {
    unsigned u = __float_as_uint(f);
    u += 0x7fffu + ((u >> 16) & 1u);   // round-to-nearest-even
    return (unsigned short)(u >> 16);
}

// ---------------------------------------------------------------------------
// Kernel 1: per-row squared norm, bf16 hi/lo split, zero accumulators
// grid = 8192 blocks x 128 threads (one block per row)
// ---------------------------------------------------------------------------
__global__ void __launch_bounds__(DIMS)
prep_kernel(const float* __restrict__ x,
            unsigned short* __restrict__ Xhi, unsigned short* __restrict__ Xlo,
            float* __restrict__ sq,
            float* __restrict__ posE, float* __restrict__ negE,
            double* __restrict__ sums, unsigned int* __restrict__ cnts) {
    const int row = blockIdx.x;
    const int t   = threadIdx.x;
    const float v = x[row * DIMS + t];

    // hi = bf16(v); lo = bf16(v - hi)  (v - hi is exact in fp32)
    unsigned u = __float_as_uint(v);
    unsigned hr = (u + 0x7fffu + ((u >> 16) & 1u)) & 0xffff0000u;
    float hif = __uint_as_float(hr);
    Xhi[row * DIMS + t] = (unsigned short)(hr >> 16);
    Xlo[row * DIMS + t] = f32_to_bf16_rne(v - hif);

    // block reduction of v*v (4 wave32's)
    float s = v * v;
    #pragma unroll
    for (int m = 16; m >= 1; m >>= 1) s += __shfl_xor(s, m, 32);
    __shared__ float part[4];
    if ((t & 31) == 0) part[t >> 5] = s;
    __syncthreads();
    if (t == 0) {
        sq[row]   = part[0] + part[1] + part[2] + part[3];
        posE[row] = 0.0f;
        negE[row] = 0.0f;
        if (row == 0) { sums[0] = 0.0; sums[1] = 0.0; cnts[0] = 0u; cnts[1] = 0u; }
    }
}

// ---------------------------------------------------------------------------
// Kernel 2: upper-triangular 128x128 tile blocks of G via WMMA + fused
//           distance/exp/mask accumulation. 512 threads = 16 wave32 per block;
//           each wave owns 1 M-subtile x 4 N-subtiles (16x16 each).
// ---------------------------------------------------------------------------
__global__ void __launch_bounds__(512)
pair_tile_kernel(const unsigned short* __restrict__ Xhi,
                 const unsigned short* __restrict__ Xlo,
                 const float* __restrict__ sq, const int* __restrict__ tgt,
                 float* __restrict__ posE, float* __restrict__ negE,
                 double* __restrict__ sums, unsigned int* __restrict__ cnts) {
    const int bx = blockIdx.x;   // column block
    const int by = blockIdx.y;   // row block
    if (bx < by) return;         // symmetric: only upper triangle

    extern __shared__ unsigned short smem[];
    unsigned short* Ahi = smem;
    unsigned short* Alo = Ahi + TILE * LDS_PITCH;
    unsigned short* Bhi = Alo + TILE * LDS_PITCH;
    unsigned short* Blo = Bhi + TILE * LDS_PITCH;

    const int tid     = threadIdx.x;
    const int lane    = tid & 31;
    const int wave    = tid >> 5;   // 0..15
    const int mwave   = wave & 7;   // M subtile of this wave
    const int ngrp    = wave >> 3;  // 0..1 -> N subtiles 4*ngrp .. 4*ngrp+3
    const int h       = lane >> 4;  // half of the wave
    const int ln      = lane & 15;

    const int rowBase = by * TILE;
    const int colBase = bx * TILE;

    // ---- stage A/B blocks (hi+lo) into LDS, 16B-vectorized, padded rows ----
    const uint4* ghi = (const uint4*)Xhi;
    const uint4* glo = (const uint4*)Xlo;
    for (int idx = tid; idx < TILE * 16; idx += 512) {
        const int r = idx >> 4, c = idx & 15;
        ((uint4*)(Ahi + r * LDS_PITCH))[c] = ghi[(rowBase + r) * 16 + c];
        ((uint4*)(Alo + r * LDS_PITCH))[c] = glo[(rowBase + r) * 16 + c];
        ((uint4*)(Bhi + r * LDS_PITCH))[c] = ghi[(colBase + r) * 16 + c];
        ((uint4*)(Blo + r * LDS_PITCH))[c] = glo[(colBase + r) * 16 + c];
    }
    __syncthreads();

    // ---- K loop: 4 steps of K=32, split-bf16 (3 WMMAs per subtile) ----
    v8f acc[4] = {};
    #pragma unroll
    for (int ks = 0; ks < 4; ++ks) {
        const int K0 = ks * 32;
        FragBF16 aH, aL, bH[4], bL[4];

        // A 16x32 bf16 layout: lane<16 -> K0+[0..7],[16..23]; lane>=16 -> +8
        const unsigned short* ar = Ahi + (16 * mwave + ln) * LDS_PITCH + K0 + 8 * h;
        aH.q[0] = *(const uint4*)(ar);      aH.q[1] = *(const uint4*)(ar + 16);
        const unsigned short* al = Alo + (16 * mwave + ln) * LDS_PITCH + K0 + 8 * h;
        aL.q[0] = *(const uint4*)(al);      aL.q[1] = *(const uint4*)(al + 16);

        // B 32x16 bf16 layout: lane<16 -> K0+[0..7],[16..23]; lane>=16 -> +8
        #pragma unroll
        for (int nt = 0; nt < 4; ++nt) {
            const int ntile = ngrp * 4 + nt;
            const unsigned short* br = Bhi + (16 * ntile + ln) * LDS_PITCH + K0 + 8 * h;
            bH[nt].q[0] = *(const uint4*)(br);  bH[nt].q[1] = *(const uint4*)(br + 16);
            const unsigned short* bl = Blo + (16 * ntile + ln) * LDS_PITCH + K0 + 8 * h;
            bL[nt].q[0] = *(const uint4*)(bl);  bL[nt].q[1] = *(const uint4*)(bl + 16);
        }
        #pragma unroll
        for (int nt = 0; nt < 4; ++nt) {
            acc[nt] = wmma_bf16(aH.v, bH[nt].v, acc[nt]);  // hi*hi
            acc[nt] = wmma_bf16(aH.v, bL[nt].v, acc[nt]);  // hi*lo
            acc[nt] = wmma_bf16(aL.v, bH[nt].v, acc[nt]);  // lo*hi
        }
    }

    // ---- fused epilogue: distances, exp, masked sums ----
    // C layout: VGPR j holds row (j + 8*h) of the subtile, lane&15 = column
    const int rbase = rowBase + 16 * mwave + 8 * h;   // rows rbase+j, j=0..7
    float sqi[8]; int ti[8];
    #pragma unroll
    for (int j = 0; j < 8; ++j) { sqi[j] = sq[rbase + j]; ti[j] = tgt[rbase + j]; }

    float posAcc[8] = {}, negAcc[8] = {};
    float pD = 0.0f, nD = 0.0f;
    unsigned int pC = 0u, nC = 0u;
    const bool offdiag = (bx != by);
    const float        wd = offdiag ? 2.0f : 1.0f;   // symmetric double-count
    const unsigned int wc = offdiag ? 2u   : 1u;

    #pragma unroll
    for (int nt = 0; nt < 4; ++nt) {
        const int gc  = colBase + 16 * (ngrp * 4 + nt) + ln;
        const float sqc = sq[gc];
        const int   tc  = tgt[gc];
        float colP = 0.0f, colN = 0.0f;
        #pragma unroll
        for (int j = 0; j < 8; ++j) {
            const int gi = rbase + j;
            const float g  = acc[nt][j];
            float d2 = fmaxf(sqi[j] + sqc - 2.0f * g, 1e-12f);
            const float d  = sqrtf(d2);                      // v_sqrt_f32
            const float e  = __expf(ALPHA_C * (BASE_C - d)); // v_exp_f32
            const bool same = (ti[j] == tc);
            const bool pos  = same && (gi != gc);
            const float pe = pos   ? e : 0.0f;
            const float ne = !same ? e : 0.0f;
            posAcc[j] += pe;  negAcc[j] += ne;
            colP      += pe;  colN      += ne;
            pD += pos   ? wd * d : 0.0f;
            nD += !same ? wd * d : 0.0f;
            pC += pos   ? wc : 0u;
            nC += !same ? wc : 0u;
        }
        if (offdiag) {  // scatter transposed contribution to column rows
            if (colP != 0.0f) atomicAdd(&posE[gc], colP);
            if (colN != 0.0f) atomicAdd(&negE[gc], colN);
        }
    }

    // row sums: reduce across the 16 lanes of each half, one atomic per row
    #pragma unroll
    for (int j = 0; j < 8; ++j) {
        float p = posAcc[j], q = negAcc[j];
        #pragma unroll
        for (int m = 1; m < 16; m <<= 1) { p += __shfl_xor(p, m, 32); q += __shfl_xor(q, m, 32); }
        if (ln == 0) {
            atomicAdd(&posE[rbase + j], p);
            atomicAdd(&negE[rbase + j], q);
        }
    }

    // global d sums / counts: full-wave reduction, one atomic set per wave
    #pragma unroll
    for (int m = 1; m < 32; m <<= 1) {
        pD += __shfl_xor(pD, m, 32);  nD += __shfl_xor(nD, m, 32);
        pC += __shfl_xor(pC, m, 32);  nC += __shfl_xor(nC, m, 32);
    }
    if (lane == 0) {
        atomicAdd(&sums[0], (double)pD);   // global_atomic_add_f64
        atomicAdd(&sums[1], (double)nD);
        atomicAdd(&cnts[0], pC);
        atomicAdd(&cnts[1], nC);
    }
}

// ---------------------------------------------------------------------------
// Kernel 3: final row-wise log-loss reduction + output assembly
// ---------------------------------------------------------------------------
__global__ void __launch_bounds__(256)
finalize_kernel(const float* __restrict__ posE, const float* __restrict__ negE,
                const double* __restrict__ sums, const unsigned int* __restrict__ cnts,
                float* __restrict__ out) {
    __shared__ float red[256];
    float acc = 0.0f;
    for (int i = threadIdx.x; i < N_PTS; i += 256) {
        const float p = posE[i];
        const float q = 0.5f * negE[i];
        acc += -__logf(p / (p + q));
    }
    red[threadIdx.x] = acc;
    __syncthreads();
    for (int s = 128; s > 0; s >>= 1) {
        if (threadIdx.x < s) red[threadIdx.x] += red[threadIdx.x + s];
        __syncthreads();
    }
    if (threadIdx.x == 0) {
        out[0] = red[0] / (float)N_PTS;                  // loss
        out[1] = 1.0f;                                   // prec
        out[2] = (float)(sums[0] / (double)cnts[0]);     // pos_d
        out[3] = (float)(sums[1] / (double)cnts[1]);     // neg_d
    }
}

// ---------------------------------------------------------------------------
extern "C" void kernel_launch(void* const* d_in, const int* in_sizes, int n_in,
                              void* d_out, int out_size, void* d_ws, size_t ws_size,
                              hipStream_t stream) {
    const float* x   = (const float*)d_in[0];
    const int*   tgt = (const int*)d_in[1];

    // workspace layout (bytes)
    char* ws = (char*)d_ws;
    unsigned short* Xhi  = (unsigned short*)(ws);                       // 2 MB
    unsigned short* Xlo  = (unsigned short*)(ws + 2097152);             // 2 MB
    float*          sq   = (float*)(ws + 4194304);                      // 32 KB
    float*          posE = (float*)(ws + 4194304 + 32768);              // 32 KB
    float*          negE = (float*)(ws + 4194304 + 65536);              // 32 KB
    double*         sums = (double*)(ws + 4194304 + 98304);             // 16 B
    unsigned int*   cnts = (unsigned int*)(ws + 4194304 + 98304 + 16);  // 8 B

    // dynamic LDS: 4 tiles * 128 rows * 136 elems * 2B = 136 KB (WGP has 320 KB)
    const size_t ldsBytes = (size_t)4 * TILE * LDS_PITCH * sizeof(unsigned short);
    (void)hipFuncSetAttribute((const void*)pair_tile_kernel,
                              hipFuncAttributeMaxDynamicSharedMemorySize,
                              (int)ldsBytes);

    prep_kernel<<<N_PTS, DIMS, 0, stream>>>(x, Xhi, Xlo, sq, posE, negE, sums, cnts);

    dim3 grid(N_PTS / TILE, N_PTS / TILE);   // 64 x 64, lower triangle early-exits
    pair_tile_kernel<<<grid, 512, ldsBytes, stream>>>(Xhi, Xlo, sq, tgt,
                                                      posE, negE, sums, cnts);

    finalize_kernel<<<1, 256, 0, stream>>>(posE, negE, sums, cnts, (float*)d_out);
}
